// IncreAudioVisualNet_41875931136661
// MI455X (gfx1250) — matrix operand order repaired and checked by reading, not compiled, not run on hardware
//
#include <hip/hip_runtime.h>
#include <hip/hip_bf16.h>

// ---------------------------------------------------------------------------
// IncreAudioVisualNet forward for MI455X (gfx1250, wave32, WMMA).
// Dominant cost: pv = tanh(visual @ Wav + bav)  [50176x768]x[768x768]
//   -> v_wmma_f32_16x16x32_bf16, fp32 accumulate.
//   B (Wav) pre-swizzled to bf16 fragment layout (one-shot 2.25MB kernel);
//   A converted on the fly, reused across 4 N-tiles per wave.
// Everything else is bandwidth-bound elementwise / small-softmax passes.
// ---------------------------------------------------------------------------

#define BB 32
#define TT 8
#define PP 196
#define DD 768
#define MM 10
#define NC 100
#define NUM_IDX 98

typedef __attribute__((ext_vector_type(16))) __bf16 v16bf;
typedef __attribute__((ext_vector_type(8)))  float  v8f;

union Bf16x16 {
    v16bf v;
    unsigned short u[16];
    uint4 q[2];
};

__device__ __forceinline__ unsigned short f2bf(float f) {
    unsigned u = __float_as_uint(f);
    u += 0x7fffu + ((u >> 16) & 1u);       // round-to-nearest-even
    return (unsigned short)(u >> 16);
}

// ---------------------------------------------------------------------------
// K0: feats2[b,m] = causal[b,m,:] . Wf2 + bf2      (B*M = 320 values)
// ---------------------------------------------------------------------------
__global__ void k_feats2(const float* __restrict__ causal,
                         const float* __restrict__ Wf2,
                         const float* __restrict__ bf2,
                         float* __restrict__ feats2) {
    int idx = blockIdx.x * blockDim.x + threadIdx.x;
    if (idx >= BB * MM) return;
    const float* c = causal + (long)idx * DD;
    float s = bf2[0];
    for (int k = 0; k < DD; ++k) s += c[k] * Wf2[k];
    feats2[idx] = s;
}

// ---------------------------------------------------------------------------
// K1: per (b,d): feats -> tanh -> softmax over M -> ci -> audio2
//     (Wf column read once, causal element is the broadcast operand)
// ---------------------------------------------------------------------------
__global__ void k_audio2(const float* __restrict__ causal,
                         const float* __restrict__ Wf,
                         const float* __restrict__ bf,
                         const float* __restrict__ feats2,
                         const float* __restrict__ audio,
                         const float* __restrict__ cparam,
                         float* __restrict__ audio2) {
    int idx = blockIdx.x * blockDim.x + threadIdx.x;   // b*DD + d
    if (idx >= BB * DD) return;
    int b = idx / DD, d = idx % DD;
    const float* cb = causal + (long)b * MM * DD;
    float fe[MM];
    #pragma unroll
    for (int m = 0; m < MM; ++m) fe[m] = bf[d];
    for (int k = 0; k < DD; ++k) {
        float wfk = Wf[(long)k * DD + d];
        #pragma unroll
        for (int m = 0; m < MM; ++m) fe[m] += cb[(long)m * DD + k] * wfk;
    }
    float mx = -1e30f;
    #pragma unroll
    for (int m = 0; m < MM; ++m) { fe[m] = tanhf(fe[m]); mx = fmaxf(mx, fe[m]); }
    float sum = 0.f;
    #pragma unroll
    for (int m = 0; m < MM; ++m) { fe[m] = expf(fe[m] - mx); sum += fe[m]; }
    float inv = 1.f / sum;
    float ci = 0.f;
    #pragma unroll
    for (int m = 0; m < MM; ++m)
        ci += feats2[b * MM + m] * (fe[m] * inv) * cb[(long)m * DD + d];
    float cp = 1.f / (1.f + expf(-cparam[0]));
    audio2[idx] = cp * ci + audio[idx];
}

// ---------------------------------------------------------------------------
// K2: pa = tanh(audio2 @ Waa + baa);  af = relu(audio2 @ Wa + ba)
// ---------------------------------------------------------------------------
__global__ void k_pa_af(const float* __restrict__ audio2,
                        const float* __restrict__ Waa, const float* __restrict__ baa,
                        const float* __restrict__ Wa,  const float* __restrict__ ba,
                        float* __restrict__ pa, float* __restrict__ af) {
    int idx = blockIdx.x * blockDim.x + threadIdx.x;
    if (idx >= BB * DD) return;
    int b = idx / DD, d = idx % DD;
    const float* a2 = audio2 + (long)b * DD;
    float saa = 0.f, sa = 0.f;
    for (int k = 0; k < DD; ++k) {
        float v = a2[k];
        saa += v * Waa[(long)k * DD + d];
        sa  += v * Wa [(long)k * DD + d];
    }
    pa[idx] = tanhf(saa + baa[d]);
    af[idx] = fmaxf(0.f, sa + ba[d]);
}

// ---------------------------------------------------------------------------
// K3a: swizzle Wav (fp32 row-major [k][n]) -> bf16 B-fragment layout:
//   Bsw[(((kk/32)*2 + hi)*768 + col)*16 + i] = bf16(Wav[(kk + hi*16 + i)*768 + col])
//   so each lane's 32x16 B fragment is one contiguous 32-byte chunk.
// ---------------------------------------------------------------------------
__global__ void k_swizzleB(const float* __restrict__ Wav,
                           unsigned short* __restrict__ Bsw) {
    int idx = blockIdx.x * blockDim.x + threadIdx.x;   // 589824 elements
    if (idx >= DD * DD) return;
    int i   = idx & 15;
    int col = (idx >> 4) % DD;
    int hi  = (idx / (16 * DD)) & 1;
    int ks  = idx / (16 * DD * 2);
    int k   = ks * 32 + hi * 16 + i;
    Bsw[idx] = f2bf(Wav[(long)k * DD + col]);
}

// ---------------------------------------------------------------------------
// K3: WMMA GEMM  pv = tanh(visual @ Wav + bav)
//   grid = (50176/128, 768/64), block = 256 (8 waves).
//   Wave w: rows [bx*128 + w*16, +16), cols [by*64, +64) as 4 16x16 tiles.
//   A fragment loaded+converted once per k-step, reused by 4 WMMAs.
// ---------------------------------------------------------------------------
__global__ void k_gemm_pv(const float* __restrict__ vis,
                          const unsigned short* __restrict__ Bsw,
                          const float* __restrict__ bav,
                          float* __restrict__ pv) {
    int lane = threadIdx.x & 31;
    int wv   = threadIdx.x >> 5;
    int m    = lane & 15;      // A row-in-tile / B,C column-in-tile
    int hi   = lane >> 4;      // lane half
    long row = (long)blockIdx.x * 128 + wv * 16 + m;      // A row (this lane)
    int  colbase = blockIdx.y * 64;

    v8f acc0 = {}, acc1 = {}, acc2 = {}, acc3 = {};
    for (int kk = 0; kk < DD; kk += 32) {
        // A fragment: lane row, K = kk + hi*16 + {0..15}
        const float* ap = vis + row * DD + kk + hi * 16;
        Bf16x16 a;
        #pragma unroll
        for (int i = 0; i < 8; ++i) {
            a.u[i]     = f2bf(ap[i]);
            a.u[8 + i] = f2bf(ap[8 + i]);
        }
        // B fragments: contiguous 32B per lane, pre-swizzled
        const uint4* bq = (const uint4*)
            (Bsw + ((long)((kk >> 5) * 2 + hi) * DD + colbase + m) * 16);
        Bf16x16 b0, b1, b2, b3;
        b0.q[0] = bq[0];      b0.q[1] = bq[1];        // cols colbase+ 0..15
        b1.q[0] = bq[32];     b1.q[1] = bq[33];       // cols colbase+16..31
        b2.q[0] = bq[64];     b2.q[1] = bq[65];       // cols colbase+32..47
        b3.q[0] = bq[96];     b3.q[1] = bq[97];       // cols colbase+48..63

        acc0 = __builtin_amdgcn_wmma_f32_16x16x32_bf16(
            false, a.v, false, b0.v, (short)0, acc0, false, false);
        acc1 = __builtin_amdgcn_wmma_f32_16x16x32_bf16(
            false, a.v, false, b1.v, (short)0, acc1, false, false);
        acc2 = __builtin_amdgcn_wmma_f32_16x16x32_bf16(
            false, a.v, false, b2.v, (short)0, acc2, false, false);
        acc3 = __builtin_amdgcn_wmma_f32_16x16x32_bf16(
            false, a.v, false, b3.v, (short)0, acc3, false, false);
    }

    long rbase = (long)blockIdx.x * 128 + wv * 16 + hi * 8;  // C: M = v + 8*hi
    #pragma unroll
    for (int nt = 0; nt < 4; ++nt) {
        v8f acc = (nt == 0) ? acc0 : (nt == 1) ? acc1 : (nt == 2) ? acc2 : acc3;
        int col = colbase + nt * 16 + m;                      // C column (lane)
        float bias = bav[col];
        #pragma unroll
        for (int v = 0; v < 8; ++v)
            pv[(rbase + v) * DD + col] = tanhf(acc[v] + bias);
    }
}

// ---------------------------------------------------------------------------
// K4: per (b,t): softmax stats over P per d, patch sums (LDS ds_add_f32),
//     rank-98 threshold (counting select), spatial mask.
// ---------------------------------------------------------------------------
__global__ void k_patch(const float* __restrict__ pv,
                        const float* __restrict__ pa,
                        float* __restrict__ msum,          // [B,T,D,2]
                        unsigned char* __restrict__ smask) // [B,T,P]
{
    __shared__ float patch[PP];
    __shared__ float thr;
    int bt = blockIdx.x;                // b*TT + t
    int b  = bt / TT;
    int tid = threadIdx.x;
    if (tid < PP) patch[tid] = 0.f;
    __syncthreads();

    const float* base = pv + (long)bt * PP * DD;
    for (int chunk = 0; chunk < DD / 256; ++chunk) {
        int d = tid + chunk * 256;
        float pab = pa[(long)b * DD + d];
        float mx = -1e30f;
        for (int p = 0; p < PP; ++p)
            mx = fmaxf(mx, base[(long)p * DD + d] * pab);
        float sum = 0.f;
        for (int p = 0; p < PP; ++p)
            sum += expf(base[(long)p * DD + d] * pab - mx);
        float inv = 1.f / sum;
        msum[((long)bt * DD + d) * 2 + 0] = mx;
        msum[((long)bt * DD + d) * 2 + 1] = sum;
        for (int p = 0; p < PP; ++p)
            atomicAdd(&patch[p], expf(base[(long)p * DD + d] * pab - mx) * inv);
    }
    __syncthreads();

    // value at ascending-sorted position NUM_IDX (counting select)
    if (tid < PP) {
        float v = patch[tid];
        int c_lt = 0, c_le = 0;
        for (int q = 0; q < PP; ++q) {
            c_lt += (patch[q] <  v);
            c_le += (patch[q] <= v);
        }
        if (c_lt <= NUM_IDX && NUM_IDX < c_le) thr = v;
    }
    __syncthreads();
    if (tid < PP)
        smask[(long)bt * PP + tid] = (patch[tid] < thr) ? 1 : 0;
}

// ---------------------------------------------------------------------------
// K5: per (b,t,d): masked reductions over P ->
//     sav = sum_p sfilt*pv ;  vpt = sum_p sfilt*vis
// ---------------------------------------------------------------------------
__global__ void k_reduce(const float* __restrict__ pv,
                         const float* __restrict__ pa,
                         const float* __restrict__ msum,
                         const unsigned char* __restrict__ smask,
                         const unsigned char* __restrict__ rmask, // bool [B,T,P,D]
                         const float* __restrict__ vis,
                         float* __restrict__ sav,
                         float* __restrict__ vpt) {
    int bt = blockIdx.x;
    int b  = bt / TT;
    int tid = threadIdx.x;
    const float* base = pv + (long)bt * PP * DD;
    const float* vbase = vis + (long)bt * PP * DD;
    const unsigned char* rbase = rmask + (long)bt * PP * DD;
    const unsigned char* sm = smask + (long)bt * PP;

    for (int chunk = 0; chunk < DD / 256; ++chunk) {
        int d = tid + chunk * 256;
        float pab = pa[(long)b * DD + d];
        float mx  = msum[((long)bt * DD + d) * 2 + 0];
        float inv = 1.f / msum[((long)bt * DD + d) * 2 + 1];
        float s_acc = 0.f, v_acc = 0.f;
        for (int p = 0; p < PP; ++p) {
            float pvv = base[(long)p * DD + d];
            float a = expf(pvv * pab - mx) * inv;
            bool drop = sm[p] && rbase[(long)p * DD + d];
            if (!drop) {
                s_acc += a * pvv;
                v_acc += a * vbase[(long)p * DD + d];
            }
        }
        sav[(long)bt * DD + d] = s_acc;
        vpt[(long)bt * DD + d] = v_acc;
    }
}

// ---------------------------------------------------------------------------
// K6: per (b,d): temporal softmax over T, vp = sum_t tattn * vpt
// ---------------------------------------------------------------------------
__global__ void k_vp(const float* __restrict__ sav,
                     const float* __restrict__ vpt,
                     const float* __restrict__ pa,
                     float* __restrict__ vp) {
    int idx = blockIdx.x * blockDim.x + threadIdx.x;
    if (idx >= BB * DD) return;
    int b = idx / DD, d = idx % DD;
    float pab = pa[idx];
    float s[TT], mx = -1e30f;
    #pragma unroll
    for (int t = 0; t < TT; ++t) {
        s[t] = sav[(long)(b * TT + t) * DD + d] * pab;
        mx = fmaxf(mx, s[t]);
    }
    float sum = 0.f;
    #pragma unroll
    for (int t = 0; t < TT; ++t) { s[t] = expf(s[t] - mx); sum += s[t]; }
    float inv = 1.f / sum;
    float acc = 0.f;
    #pragma unroll
    for (int t = 0; t < TT; ++t)
        acc += s[t] * inv * vpt[(long)(b * TT + t) * DD + d];
    vp[idx] = acc;
}

// ---------------------------------------------------------------------------
// K7: vf = relu(vp @ Wv + bv)
// ---------------------------------------------------------------------------
__global__ void k_vf(const float* __restrict__ vp,
                     const float* __restrict__ Wv, const float* __restrict__ bv,
                     float* __restrict__ vf) {
    int idx = blockIdx.x * blockDim.x + threadIdx.x;
    if (idx >= BB * DD) return;
    int b = idx / DD, d = idx % DD;
    const float* v = vp + (long)b * DD;
    float s = 0.f;
    for (int k = 0; k < DD; ++k) s += v[k] * Wv[(long)k * DD + d];
    vf[idx] = fmaxf(0.f, s + bv[d]);
}

// ---------------------------------------------------------------------------
// K8: logits = (af + vf) @ Wc + bc
// ---------------------------------------------------------------------------
__global__ void k_logits(const float* __restrict__ af,
                         const float* __restrict__ vf,
                         const float* __restrict__ Wc, const float* __restrict__ bc,
                         float* __restrict__ out) {
    int idx = blockIdx.x * blockDim.x + threadIdx.x;
    if (idx >= BB * NC) return;
    int b = idx / NC, c = idx % NC;
    const float* a = af + (long)b * DD;
    const float* v = vf + (long)b * DD;
    float s = bc[c];
    for (int k = 0; k < DD; ++k) s += (a[k] + v[k]) * Wc[(long)k * NC + c];
    out[idx] = s;
}

// ---------------------------------------------------------------------------
extern "C" void kernel_launch(void* const* d_in, const int* in_sizes, int n_in,
                              void* d_out, int out_size, void* d_ws, size_t ws_size,
                              hipStream_t stream) {
    (void)in_sizes; (void)n_in; (void)out_size; (void)ws_size;

    const float* visual  = (const float*)d_in[0];
    const float* audio   = (const float*)d_in[1];
    const float* causal  = (const float*)d_in[2];
    const unsigned char* rmask = (const unsigned char*)d_in[3];   // bool
    const float* Wa  = (const float*)d_in[4];
    const float* ba  = (const float*)d_in[5];
    const float* Wv  = (const float*)d_in[6];
    const float* bv  = (const float*)d_in[7];
    const float* Waa = (const float*)d_in[8];
    const float* baa = (const float*)d_in[9];
    const float* Wav = (const float*)d_in[10];
    const float* bav = (const float*)d_in[11];
    const float* Wf  = (const float*)d_in[12];
    const float* bf  = (const float*)d_in[13];
    const float* Wf2 = (const float*)d_in[14];
    const float* bf2 = (const float*)d_in[15];
    const float* Wc  = (const float*)d_in[16];
    const float* bc  = (const float*)d_in[17];
    const float* cparam = (const float*)d_in[18];
    float* out = (float*)d_out;

    // workspace layout: Bsw first (32B-aligned fragments), then floats
    unsigned short* Bsw = (unsigned short*)d_ws;            // DD*DD bf16
    float* pv     = (float*)((char*)d_ws + (long)DD * DD * 2);
    float* msum   = pv     + (long)BB * TT * PP * DD;       // B*T*D*2
    float* sav    = msum   + (long)BB * TT * DD * 2;        // B*T*D
    float* vpt    = sav    + (long)BB * TT * DD;            // B*T*D
    float* audio2 = vpt    + (long)BB * TT * DD;            // B*D
    float* pa     = audio2 + (long)BB * DD;
    float* af     = pa     + (long)BB * DD;
    float* vp     = af     + (long)BB * DD;
    float* vf     = vp     + (long)BB * DD;
    float* feats2 = vf     + (long)BB * DD;                 // B*M
    unsigned char* smask = (unsigned char*)(feats2 + BB * MM); // B*T*P bytes

    k_feats2<<<(BB * MM + 63) / 64, 64, 0, stream>>>(causal, Wf2, bf2, feats2);
    k_audio2<<<(BB * DD) / 256, 256, 0, stream>>>(causal, Wf, bf, feats2, audio,
                                                  cparam, audio2);
    k_pa_af<<<(BB * DD) / 256, 256, 0, stream>>>(audio2, Waa, baa, Wa, ba, pa, af);

    k_swizzleB<<<(DD * DD) / 256, 256, 0, stream>>>(Wav, Bsw);

    dim3 gg(BB * TT * PP / 128, DD / 64, 1);                // (392, 12)
    k_gemm_pv<<<gg, 256, 0, stream>>>(visual, Bsw, bav, pv);

    k_patch<<<BB * TT, 256, 0, stream>>>(pv, pa, msum, smask);
    k_reduce<<<BB * TT, 256, 0, stream>>>(pv, pa, msum, smask, rmask, visual,
                                          sav, vpt);
    k_vp<<<(BB * DD) / 256, 256, 0, stream>>>(sav, vpt, pa, vp);
    k_vf<<<(BB * DD) / 256, 256, 0, stream>>>(vp, Wv, bv, vf);
    k_logits<<<(BB * NC + 255) / 256, 256, 0, stream>>>(af, vf, Wc, bc, out);
}
